// Attention_18588618457429
// MI455X (gfx1250) — matrix operand order repaired and checked by reading, not compiled, not run on hardware
//
#include <hip/hip_runtime.h>
#include <hip/hip_bf16.h>

// MHA for MI455X (gfx1250, wave32). Pipeline:
//   1) qkv_gemm:  x[8192,768] @ w_qkv^T -> bf16 Q,K,V arranged [B,H,N,64]
//   2) flash_attn: online-softmax attention, WMMA bf16, O stored bf16 [B*N,768]
//   3) proj_gemm: O @ w_proj^T + b -> fp32 d_out (non-temporal stores)
// Workspace: 4 bf16 buffers of B*N*768 = 6,291,456 elems each (~48 MB total).

#define NH   12
#define HD   64
#define CDIM 768
#define SEQ  2048
#define BATCH 4

typedef __attribute__((ext_vector_type(16))) __bf16        v16bf;
typedef __attribute__((ext_vector_type(8)))  float         v8f;
typedef __attribute__((ext_vector_type(8)))  unsigned int  v8u;

__device__ __forceinline__ unsigned short f32_to_bf16(float f) {
    unsigned int u = __builtin_bit_cast(unsigned int, f);
    u += 0x7FFFu + ((u >> 16) & 1u);           // round-to-nearest-even
    return (unsigned short)(u >> 16);
}
__device__ __forceinline__ unsigned int pack_bf16(float lo, float hi) {
    return (unsigned int)f32_to_bf16(lo) | ((unsigned int)f32_to_bf16(hi) << 16);
}
__device__ __forceinline__ v8f wmma_bf16f32(v8u a, v8u b, v8f c) {
    return __builtin_amdgcn_wmma_f32_16x16x32_bf16(
        false, __builtin_bit_cast(v16bf, a),
        false, __builtin_bit_cast(v16bf, b),
        (short)0, c, false, false);
}
// 16-bit A-matrix 16x32 K index for VGPR t (ISA 7.12.2): lanes<16 hold K{0..7,16..23}
__device__ __forceinline__ int a_kidx(int t, int hi8) {
    return ((t < 4) ? (2 * t) : (16 + 2 * (t - 4))) + hi8;
}

// ---------------------------------------------------------------------------
// Kernel 1: QKV GEMM. C[m,o] = sum_k x[m,k] * w_qkv[o,k]; scatter to Q/K/V.
// Block: 256 thr (8 waves) computes 64(M) x 128(N); wave = 16(M) x 64(N).
// ---------------------------------------------------------------------------
__global__ __launch_bounds__(256) void qkv_gemm_kernel(
    const float* __restrict__ x, const float* __restrict__ wq,
    unsigned short* __restrict__ qb, unsigned short* __restrict__ kb,
    unsigned short* __restrict__ vb)
{
    __shared__ unsigned int As[64 * 18];    // 64 rows x 32 bf16 (16 u32 + 2 pad)
    __shared__ unsigned int Bs[128 * 18];   // 128 rows x 32 bf16

    const int tid = threadIdx.x;
    const int lane = tid & 31;
    const int wv = tid >> 5;
    const int lm = lane & 15, hi = lane >> 4, hi8 = hi * 8;
    const int wm = wv & 3, wn = wv >> 2;
    const int m_blk = blockIdx.x * 64;
    const int n_blk = blockIdx.y * 128;

    v8f acc[4];
    #pragma unroll
    for (int j = 0; j < 4; ++j)
        #pragma unroll
        for (int r = 0; r < 8; ++r) acc[j][r] = 0.0f;

    for (int k0 = 0; k0 < CDIM; k0 += 32) {
        __syncthreads();
        #pragma unroll
        for (int i = 0; i < 2; ++i) {           // x tile: 64x32 f32 -> bf16 LDS
            int idx = tid * 2 + i;
            int row = idx >> 3, c4 = idx & 7;
            float4 f = *(const float4*)(x + (size_t)(m_blk + row) * CDIM + k0 + c4 * 4);
            As[row * 18 + c4 * 2 + 0] = pack_bf16(f.x, f.y);
            As[row * 18 + c4 * 2 + 1] = pack_bf16(f.z, f.w);
        }
        #pragma unroll
        for (int i = 0; i < 4; ++i) {           // w tile: 128x32 f32 -> bf16 LDS
            int idx = tid * 4 + i;
            int row = idx >> 3, c4 = idx & 7;
            float4 f = *(const float4*)(wq + (size_t)(n_blk + row) * CDIM + k0 + c4 * 4);
            Bs[row * 18 + c4 * 2 + 0] = pack_bf16(f.x, f.y);
            Bs[row * 18 + c4 * 2 + 1] = pack_bf16(f.z, f.w);
        }
        if (k0 + 32 < CDIM) {                   // prefetch next tiles under WMMA
            int idx = tid * 2;
            __builtin_prefetch(x + (size_t)(m_blk + (idx >> 3)) * CDIM + k0 + 32 + (idx & 7) * 4, 0, 0);
            idx = tid * 4;
            __builtin_prefetch(wq + (size_t)(n_blk + (idx >> 3)) * CDIM + k0 + 32 + (idx & 7) * 4, 0, 0);
        }
        __syncthreads();

        v8u af;
        #pragma unroll
        for (int t = 0; t < 8; ++t)
            af[t] = As[(wm * 16 + lm) * 18 + (a_kidx(t, hi8) >> 1)];
        #pragma unroll
        for (int j = 0; j < 4; ++j) {
            v8u bf;
            #pragma unroll
            for (int t = 0; t < 8; ++t)         // B layout: K = a_kidx(t, hi8)
                bf[t] = Bs[(wn * 64 + j * 16 + lm) * 18 + (a_kidx(t, hi8) >> 1)];
            acc[j] = wmma_bf16f32(af, bf, acc[j]);
        }
    }

    // Epilogue: o = s*768 + h*64 + d; scatter bf16 into [B,H,N,64] Q/K/V.
    #pragma unroll
    for (int j = 0; j < 4; ++j) {
        int o = n_blk + wn * 64 + j * 16 + lm;
        int s = o / CDIM;
        int rem = o - s * CDIM;
        int h = rem >> 6, d = rem & 63;
        unsigned short* dst = (s == 0) ? qb : ((s == 1) ? kb : vb);
        #pragma unroll
        for (int r = 0; r < 8; ++r) {
            int m = m_blk + wm * 16 + r + hi8;
            int bb = m >> 11, n = m & 2047;
            dst[((size_t)((bb * NH + h) * SEQ + n) << 6) + d] = f32_to_bf16(acc[j][r]);
        }
    }
}

// ---------------------------------------------------------------------------
// Kernel 2: Flash attention. Block = (b,h, 128 Q rows); wave = 16 Q rows.
// S^T = K.Q^T (keys as M, queries as N) so each query's 32 scores live in a
// lane pair (q, q+16); O^T = V^T.P^T keeps alpha/l rescale lane-uniform.
// ---------------------------------------------------------------------------
__global__ __launch_bounds__(256) void flash_attn_kernel(
    const unsigned short* __restrict__ q, const unsigned short* __restrict__ k,
    const unsigned short* __restrict__ v, unsigned short* __restrict__ o)
{
    __shared__ unsigned int   Ks[32 * 34];   // 32 keys x 64 bf16 (32 u32 + 2 pad)
    __shared__ unsigned short Vt[64 * 34];   // transposed: 64 d x 32 keys (+2 pad)

    const int tid = threadIdx.x;
    const int lane = tid & 31;
    const int wv = tid >> 5;
    const int lm = lane & 15, hi = lane >> 4, hi8 = hi * 8;
    const int h = blockIdx.y, b = blockIdx.z;
    const int bh = b * NH + h;
    const unsigned short* qh = q + (size_t)bh * SEQ * HD;
    const unsigned short* kh = k + (size_t)bh * SEQ * HD;
    const unsigned short* vh = v + (size_t)bh * SEQ * HD;
    const int qrow = blockIdx.x * 128 + wv * 16 + lm;

    // Q B-fragments (held all kernel): B[kd][n] = Q[q0+n][chunk*32 + kd]
    v8u qf[2];
    #pragma unroll
    for (int c = 0; c < 2; ++c) {
        const unsigned int* p = (const unsigned int*)(qh + (size_t)qrow * HD + c * 32 + hi * 8);
        #pragma unroll
        for (int t = 0; t < 8; ++t) qf[c][t] = p[(t < 4) ? t : (t + 4)];
    }

    v8f oacc[4];
    #pragma unroll
    for (int j = 0; j < 4; ++j)
        #pragma unroll
        for (int r = 0; r < 8; ++r) oacc[j][r] = 0.0f;

    float m_run = -1e30f, l_run = 0.0f;
    const float SC2 = 0.18033688011f;        // (1/sqrt(64)) * log2(e)
    const int row = tid >> 3, g = tid & 7;   // cooperative tile loaders

    for (int kb0 = 0; kb0 < SEQ; kb0 += 32) {
        __syncthreads();
        {   // K tile row-major; V tile transposed into Vt at store time
            uint4 kk = *(const uint4*)(kh + (size_t)(kb0 + row) * HD + g * 8);
            Ks[row * 34 + g * 4 + 0] = kk.x;
            Ks[row * 34 + g * 4 + 1] = kk.y;
            Ks[row * 34 + g * 4 + 2] = kk.z;
            Ks[row * 34 + g * 4 + 3] = kk.w;
            uint4 vvq = *(const uint4*)(vh + (size_t)(kb0 + row) * HD + g * 8);
            unsigned int vw[4] = {vvq.x, vvq.y, vvq.z, vvq.w};
            #pragma unroll
            for (int i = 0; i < 4; ++i) {
                Vt[(g * 8 + 2 * i + 0) * 34 + row] = (unsigned short)(vw[i] & 0xFFFFu);
                Vt[(g * 8 + 2 * i + 1) * 34 + row] = (unsigned short)(vw[i] >> 16);
            }
            if (kb0 + 32 < SEQ) {            // prefetch next K/V tiles under compute
                __builtin_prefetch(kh + (size_t)(kb0 + 32 + row) * HD + g * 8, 0, 0);
                __builtin_prefetch(vh + (size_t)(kb0 + 32 + row) * HD + g * 8, 0, 0);
            }
        }
        __syncthreads();

        // S^T tiles: tile0 = keys kb0..+15, tile1 = keys kb0+16..+31
        v8f st[2];
        #pragma unroll
        for (int tile = 0; tile < 2; ++tile) {
            v8f s;
            #pragma unroll
            for (int r = 0; r < 8; ++r) s[r] = 0.0f;
            #pragma unroll
            for (int c = 0; c < 2; ++c) {
                v8u af;
                #pragma unroll
                for (int t = 0; t < 8; ++t)
                    af[t] = Ks[(tile * 16 + lm) * 34 + ((c * 32 + a_kidx(t, hi8)) >> 1)];
                s = wmma_bf16f32(af, qf[c], s);
            }
            st[tile] = s;
        }

        // Online softmax: query q = lane%16; halves combine via shfl_xor(16).
        float ts0[8], ts1[8];
        float mloc = -1e30f;
        #pragma unroll
        for (int r = 0; r < 8; ++r) {
            ts0[r] = st[0][r] * SC2;
            ts1[r] = st[1][r] * SC2;
            mloc = fmaxf(mloc, fmaxf(ts0[r], ts1[r]));
        }
        mloc = fmaxf(mloc, __shfl_xor(mloc, 16, 32));
        float mnew = fmaxf(m_run, mloc);
        float alpha = exp2f(m_run - mnew);
        float U[8], Vv[8];
        float ssum = 0.0f;
        #pragma unroll
        for (int r = 0; r < 8; ++r) {
            U[r]  = exp2f(ts0[r] - mnew);
            Vv[r] = exp2f(ts1[r] - mnew);
            ssum += U[r] + Vv[r];
        }
        ssum += __shfl_xor(ssum, 16, 32);
        l_run = l_run * alpha + ssum;
        m_run = mnew;
        #pragma unroll
        for (int j = 0; j < 4; ++j)
            #pragma unroll
            for (int r = 0; r < 8; ++r) oacc[j][r] *= alpha;

        // Build P^T B-fragment from C-layout registers (partner via shfl_xor 16).
        float pU[8], pV[8];
        #pragma unroll
        for (int r = 0; r < 8; ++r) {
            pU[r] = __shfl_xor(U[r], 16, 32);
            pV[r] = __shfl_xor(Vv[r], 16, 32);
        }
        v8u pf;
        #pragma unroll
        for (int t = 0; t < 4; ++t) {        // lanes<16: keys 0..7; lanes>=16: 16..23
            float lo = hi ? pV[2 * t]     : U[2 * t];
            float hh = hi ? pV[2 * t + 1] : U[2 * t + 1];
            pf[t] = pack_bf16(lo, hh);
        }
        #pragma unroll
        for (int t = 4; t < 8; ++t) {        // lanes<16: keys 8..15; lanes>=16: 24..31
            int s2 = 2 * (t - 4);
            float lo = hi ? Vv[s2]     : pU[s2];
            float hh = hi ? Vv[s2 + 1] : pU[s2 + 1];
            pf[t] = pack_bf16(lo, hh);
        }

        // O^T += V^T x P^T : A[m][kk] = V[kb0+kk][16j + m]
        #pragma unroll
        for (int j = 0; j < 4; ++j) {
            v8u vf;
            int d = j * 16 + lm;
            #pragma unroll
            for (int t = 0; t < 8; ++t) {
                int kk = 2 * t + 2 * hi8;    // even -> 4B-aligned u32 read
                vf[t] = *(const unsigned int*)&Vt[d * 34 + kk];
            }
            oacc[j] = wmma_bf16f32(vf, pf, oacc[j]);
        }
    }

    float invl = 1.0f / l_run;               // per-lane query is fixed (lm)
    #pragma unroll
    for (int j = 0; j < 4; ++j)
        #pragma unroll
        for (int r = 0; r < 8; ++r) {
            int d = j * 16 + r + hi8;
            o[(size_t)(b * SEQ + qrow) * CDIM + h * HD + d] =
                f32_to_bf16(oacc[j][r] * invl);
        }
}

// ---------------------------------------------------------------------------
// Kernel 3: output projection. out[m,o] = sum_c a[m,c]*w_proj[o,c] + bias[o].
// ---------------------------------------------------------------------------
__global__ __launch_bounds__(256) void proj_gemm_kernel(
    const unsigned short* __restrict__ a, const float* __restrict__ wp,
    const float* __restrict__ bias, float* __restrict__ out)
{
    __shared__ unsigned int As[64 * 18];
    __shared__ unsigned int Bs[128 * 18];

    const int tid = threadIdx.x;
    const int lane = tid & 31;
    const int wv = tid >> 5;
    const int lm = lane & 15, hi = lane >> 4, hi8 = hi * 8;
    const int wm = wv & 3, wn = wv >> 2;
    const int m_blk = blockIdx.x * 64;
    const int n_blk = blockIdx.y * 128;

    v8f acc[4];
    #pragma unroll
    for (int j = 0; j < 4; ++j)
        #pragma unroll
        for (int r = 0; r < 8; ++r) acc[j][r] = 0.0f;

    const unsigned int* au = (const unsigned int*)a;

    for (int k0 = 0; k0 < CDIM; k0 += 32) {
        __syncthreads();
        #pragma unroll
        for (int i = 0; i < 4; ++i) {        // a tile: 64x32 bf16 direct copy
            int idx = tid * 4 + i;
            int row = idx >> 4, c2 = idx & 15;
            As[row * 18 + c2] = au[(((size_t)(m_blk + row) * CDIM + k0) >> 1) + c2];
        }
        #pragma unroll
        for (int i = 0; i < 4; ++i) {        // w_proj tile: 128x32 f32 -> bf16
            int idx = tid * 4 + i;
            int row = idx >> 3, c4 = idx & 7;
            float4 f = *(const float4*)(wp + (size_t)(n_blk + row) * CDIM + k0 + c4 * 4);
            Bs[row * 18 + c4 * 2 + 0] = pack_bf16(f.x, f.y);
            Bs[row * 18 + c4 * 2 + 1] = pack_bf16(f.z, f.w);
        }
        if (k0 + 32 < CDIM) {                // prefetch next tiles under WMMA
            int idx = tid * 4;
            __builtin_prefetch((const char*)(au + (((size_t)(m_blk + (idx >> 4)) * CDIM + k0 + 32) >> 1) + (idx & 15)), 0, 0);
            __builtin_prefetch(wp + (size_t)(n_blk + (idx >> 3)) * CDIM + k0 + 32 + (idx & 7) * 4, 0, 0);
        }
        __syncthreads();

        v8u af;
        #pragma unroll
        for (int t = 0; t < 8; ++t)
            af[t] = As[(wm * 16 + lm) * 18 + (a_kidx(t, hi8) >> 1)];
        #pragma unroll
        for (int j = 0; j < 4; ++j) {
            v8u bf;
            #pragma unroll
            for (int t = 0; t < 8; ++t)
                bf[t] = Bs[(wn * 64 + j * 16 + lm) * 18 + (a_kidx(t, hi8) >> 1)];
            acc[j] = wmma_bf16f32(af, bf, acc[j]);
        }
    }

    #pragma unroll
    for (int j = 0; j < 4; ++j) {
        int oc = n_blk + wn * 64 + j * 16 + lm;
        float bv = bias[oc];
        #pragma unroll
        for (int r = 0; r < 8; ++r) {
            int m = m_blk + wm * 16 + r + hi8;
            // Final output: written once, never re-read -> non-temporal (TH=NT)
            __builtin_nontemporal_store(acc[j][r] + bv, out + (size_t)m * CDIM + oc);
        }
    }
}

// ---------------------------------------------------------------------------
extern "C" void kernel_launch(void* const* d_in, const int* in_sizes, int n_in,
                              void* d_out, int out_size, void* d_ws, size_t ws_size,
                              hipStream_t stream) {
    const float* x      = (const float*)d_in[0];
    const float* w_qkv  = (const float*)d_in[1];
    const float* w_proj = (const float*)d_in[2];
    const float* b_proj = (const float*)d_in[3];
    float* out = (float*)d_out;

    const size_t per = (size_t)BATCH * NH * SEQ * HD;   // 6,291,456 bf16 elems
    unsigned short* qb = (unsigned short*)d_ws;
    unsigned short* kb = qb + per;
    unsigned short* vb = kb + per;
    unsigned short* ab = vb + per;

    qkv_gemm_kernel<<<dim3(128, 18), 256, 0, stream>>>(x, w_qkv, qb, kb, vb);
    flash_attn_kernel<<<dim3(16, NH, BATCH), 256, 0, stream>>>(qb, kb, vb, ab);
    proj_gemm_kernel<<<dim3(128, 6), 256, 0, stream>>>(ab, w_proj, b_proj, out);

    (void)in_sizes; (void)n_in; (void)out_size; (void)ws_size;
}
